// BottleneckAwareGate_40029095199048
// MI455X (gfx1250) — matrix operand order repaired and checked by reading, not compiled, not run on hardware
//
#include <hip/hip_runtime.h>
#include <hip/hip_bf16.h>

// ---------------------------------------------------------------------------
// BottleneckAwareGate on MI455X (gfx1250, wave32, WMMA bf16)
//   GEMM1: [B,512] x [512,128]  -> v_wmma_f32_16x16x32_bf16, 16 k-steps x 8 n-tiles
//   LN    : per-row over 128 via __shfl_xor butterflies (D-layout aware)
//   GEMM2: [B,128] x [128,64]   -> WMMA, A-frags re-read from LDS (transpose)
//   GEMM3+softmax: per-lane fp32 (1% of FLOPs)
// Weights pre-packed once per launch into WMMA B-fragment order (bf16) in d_ws.
// fp32->bf16 now uses hardware packed converts (v_cvt_pk_bf16_f32) instead of
// a manual RNE bit-sequence, cutting the VALU overhead in the GEMM1 hot loop.
// ---------------------------------------------------------------------------

typedef __attribute__((ext_vector_type(16))) __bf16    v16bf;
typedef __attribute__((ext_vector_type(2)))  __bf16    v2bf;
typedef __attribute__((ext_vector_type(8)))  float     v8f;
typedef __attribute__((ext_vector_type(2)))  float     v2f;
typedef __attribute__((ext_vector_type(8)))  unsigned  v8u;

#define LN_EPS 1e-5f

static __device__ __forceinline__ unsigned pk2(float lo, float hi) {
  // hardware packed fp32 -> 2 x bf16 (RNE): v_cvt_pk_bf16_f32
  v2f  v = {lo, hi};
  v2bf b = __builtin_convertvector(v, v2bf);
  return __builtin_bit_cast(unsigned, b);
}
static __device__ __forceinline__ unsigned short f2bf(float f) {
  __bf16 b = (__bf16)f;                    // hardware v_cvt_bf16_f32
  return __builtin_bit_cast(unsigned short, b);
}

// ---------------------------------------------------------------------------
// Pack W1 (512x128) and W2 (128x64) fp32 row-major -> bf16 WMMA B-fragments.
// B-matrix 32x16 bf16 layout: lane L<16 holds col N=L, K=0..7 in dwords 0-3,
// K=16..23 in dwords 4-7; lanes 16-31 hold same cols, K=8..15 and K=24..31.
// Fragment storage: frag[tile][lane][8 dwords], tile = kk*NT + nt.
// W1 frags: v8u[0 .. 4095]  (16 ksteps x 8 ntiles x 32 lanes)
// W2 frags: v8u[4096 .. 4607] ( 4 ksteps x 4 ntiles x 32 lanes)
// ---------------------------------------------------------------------------
__global__ void __launch_bounds__(256)
gate_pack_weights(const float* __restrict__ W1, const float* __restrict__ W2,
                  unsigned* __restrict__ ws) {
  int u = blockIdx.x * 256 + threadIdx.x;
  if (u < 4096) {                       // W1: K=512 (16 ksteps), N=128 (8 ntiles)
    int lane = u & 31, tile = u >> 5;
    int kk = tile >> 3, nt = tile & 7;
    int n  = nt * 16 + (lane & 15);
    int kb = kk * 32 + (lane >> 4) * 8;
    unsigned* dst = ws + (size_t)u * 8;
#pragma unroll
    for (int j = 0; j < 8; ++j)
      dst[j] = pk2(W1[(kb + 2*j + (j & 4) * 2) * 128 + n], W1[(kb + 2*j + (j & 4) * 2 + 1) * 128 + n]);
  } else if (u < 4608) {                // W2: K=128 (4 ksteps), N=64 (4 ntiles)
    int v = u - 4096;
    int lane = v & 31, tile = v >> 5;
    int kk = tile >> 2, nt = tile & 3;
    int n  = nt * 16 + (lane & 15);
    int kb = kk * 32 + (lane >> 4) * 8;
    unsigned* dst = ws + 32768 + (size_t)v * 8;
#pragma unroll
    for (int j = 0; j < 8; ++j)
      dst[j] = pk2(W2[(kb + 2*j + (j & 4) * 2) * 64 + n], W2[(kb + 2*j + (j & 4) * 2 + 1) * 64 + n]);
  }
}

// ---------------------------------------------------------------------------
// Main kernel: 256 threads = 8 waves; each wave owns 16 rows.
// ---------------------------------------------------------------------------
__global__ void __launch_bounds__(256)
gate_main(const float* __restrict__ lstm, const float* __restrict__ bot,
          const unsigned* __restrict__ wfrag,
          const float* __restrict__ b1, const float* __restrict__ gam,
          const float* __restrict__ bet, const float* __restrict__ b2,
          const float* __restrict__ W3, const float* __restrict__ b3,
          float* __restrict__ out, int B) {
  __shared__ unsigned short hbuf[8 * 16 * 128];   // 32 KB: per-wave 16x128 bf16

  const int tid  = threadIdx.x;
  const int wave = tid >> 5;
  const int lane = tid & 31;
  const int ln   = lane & 15;       // column-within-tile / row-within-A-frag
  const int hi   = lane >> 4;       // half-wave selector
  const int rowBase = blockIdx.x * 128 + wave * 16;
  if (rowBase >= B) return;

  int arow = rowBase + ln;
  if (arow >= B) arow = B - 1;      // clamp loads for (unused) partial tiles
  const size_t lrow = (size_t)arow * 448;
  const size_t brow = (size_t)arow * 64;
  const int    off  = hi * 8;       // K sub-chunk per 16-bit A layout

  // ---------------- GEMM1: 512 -> 128, bf16 WMMA ----------------
  v8f acc[8];
  const v8f vz = {0.f,0.f,0.f,0.f,0.f,0.f,0.f,0.f};
#pragma unroll
  for (int t = 0; t < 8; ++t) acc[t] = vz;

  for (int kk = 0; kk < 16; ++kk) {
    const float* src; int col;
    if (kk < 14) { src = lstm + lrow; col = kk * 32 + off; }         // lstm part
    else         { src = bot  + brow; col = (kk - 14) * 32 + off; }  // bottleneck
    float4 x0 = *(const float4*)(src + col);
    float4 x1 = *(const float4*)(src + col + 4);
    float4 x2 = *(const float4*)(src + col + 16);
    float4 x3 = *(const float4*)(src + col + 20);
    v8u au;
    au[0] = pk2(x0.x, x0.y); au[1] = pk2(x0.z, x0.w);
    au[2] = pk2(x1.x, x1.y); au[3] = pk2(x1.z, x1.w);
    au[4] = pk2(x2.x, x2.y); au[5] = pk2(x2.z, x2.w);
    au[6] = pk2(x3.x, x3.y); au[7] = pk2(x3.z, x3.w);
    v16bf av = __builtin_bit_cast(v16bf, au);

    const v8u* wp = (const v8u*)wfrag + (size_t)kk * 8 * 32 + lane;
#pragma unroll
    for (int nt = 0; nt < 8; ++nt) {
      v16bf bv = __builtin_bit_cast(v16bf, wp[nt * 32]);
      acc[nt] = __builtin_amdgcn_wmma_f32_16x16x32_bf16(
          false, av, false, bv, (short)0, acc[nt], false, false);
    }
  }

  // ---------------- bias + ReLU + LayerNorm ----------------
  // D layout: lane half h holds rows m+8*h in VGPR m, col = 16*t + ln.
  float bia[8], ga[8], be[8];
#pragma unroll
  for (int t = 0; t < 8; ++t) {
    bia[t] = b1 [t * 16 + ln];
    ga [t] = gam[t * 16 + ln];
    be [t] = bet[t * 16 + ln];
  }
  float mu[8], rs[8];
#pragma unroll
  for (int m = 0; m < 8; ++m) {
    float s = 0.f;
#pragma unroll
    for (int t = 0; t < 8; ++t) {
      float x = acc[t][m] + bia[t];
      x = fmaxf(x, 0.f);
      acc[t][m] = x;
      s += x;
    }
    s += __shfl_xor(s, 1); s += __shfl_xor(s, 2);
    s += __shfl_xor(s, 4); s += __shfl_xor(s, 8);
    mu[m] = s * (1.0f / 128.0f);
  }
#pragma unroll
  for (int m = 0; m < 8; ++m) {
    float q = 0.f;
#pragma unroll
    for (int t = 0; t < 8; ++t) { float d = acc[t][m] - mu[m]; q += d * d; }
    q += __shfl_xor(q, 1); q += __shfl_xor(q, 2);
    q += __shfl_xor(q, 4); q += __shfl_xor(q, 8);
    rs[m] = rsqrtf(q * (1.0f / 128.0f) + LN_EPS);
  }

  // transpose normalized h (bf16) into LDS, row-major [16][128] per wave
  unsigned short* hb = hbuf + wave * 2048;
#pragma unroll
  for (int m = 0; m < 8; ++m) {
    const int row = m + hi * 8;
#pragma unroll
    for (int t = 0; t < 8; ++t) {
      float y = (acc[t][m] - mu[m]) * rs[m] * ga[t] + be[t];
      hb[row * 128 + t * 16 + ln] = f2bf(y);
    }
  }
  asm volatile("s_wait_dscnt 0" ::: "memory");   // stores visible before frag reads

  // ---------------- GEMM2: 128 -> 64, bf16 WMMA ----------------
  v8f acc2[4];
#pragma unroll
  for (int t = 0; t < 4; ++t) acc2[t] = vz;

#pragma unroll
  for (int kk = 0; kk < 4; ++kk) {
    const unsigned short* hp = hb + ln * 128 + kk * 32 + hi * 8;
    uint4 lo = *(const uint4*)hp;          // K = kb..kb+7
    uint4 h4 = *(const uint4*)(hp + 16);   // K = kb+16..kb+23
    v8u au;
    au[0] = lo.x; au[1] = lo.y; au[2] = lo.z; au[3] = lo.w;
    au[4] = h4.x; au[5] = h4.y; au[6] = h4.z; au[7] = h4.w;
    v16bf av = __builtin_bit_cast(v16bf, au);

    const v8u* wp = (const v8u*)wfrag + 4096 + (size_t)kk * 4 * 32 + lane;
#pragma unroll
    for (int nt = 0; nt < 4; ++nt) {
      v16bf bv = __builtin_bit_cast(v16bf, wp[nt * 32]);
      acc2[nt] = __builtin_amdgcn_wmma_f32_16x16x32_bf16(
          false, av, false, bv, (short)0, acc2[nt], false, false);
    }
  }
  asm volatile("s_wait_dscnt 0" ::: "memory");   // frag reads done before reuse

  // ---------------- bias + ReLU, spill h2 fp32 to LDS ----------------
  float* h2 = (float*)hb;                   // reuse per-wave 4 KB region
#pragma unroll
  for (int m = 0; m < 8; ++m) {
    const int row = m + hi * 8;
#pragma unroll
    for (int t = 0; t < 4; ++t) {
      float x = acc2[t][m] + b2[t * 16 + ln];
      x = fmaxf(x, 0.f);
      h2[row * 64 + t * 16 + ln] = x;
    }
  }
  asm volatile("s_wait_dscnt 0" ::: "memory");

  // ---------------- GEMM3 (64 -> 8) + softmax ----------------
  // lane = (row ln, expert-half hi): lanes 0-15 experts 0..3, 16-31 experts 4..7
  const int ebase = hi * 4;
  float lg0 = b3[ebase + 0], lg1 = b3[ebase + 1];
  float lg2 = b3[ebase + 2], lg3 = b3[ebase + 3];
  const float* h2r = h2 + ln * 64;
#pragma unroll 8
  for (int c = 0; c < 64; ++c) {
    float  hv = h2r[c];
    float4 w  = *(const float4*)(W3 + c * 8 + ebase);
    lg0 = fmaf(hv, w.x, lg0); lg1 = fmaf(hv, w.y, lg1);
    lg2 = fmaf(hv, w.z, lg2); lg3 = fmaf(hv, w.w, lg3);
  }
  float mx = fmaxf(fmaxf(lg0, lg1), fmaxf(lg2, lg3));
  mx = fmaxf(mx, __shfl_xor(mx, 16));
  float e0 = __expf(lg0 - mx), e1 = __expf(lg1 - mx);
  float e2 = __expf(lg2 - mx), e3 = __expf(lg3 - mx);
  float s = e0 + e1 + e2 + e3;
  s += __shfl_xor(s, 16);
  float inv = 1.0f / s;

  const int orow = rowBase + ln;
  if (orow < B) {
    float4 res = { e0 * inv, e1 * inv, e2 * inv, e3 * inv };
    *(float4*)(out + (size_t)orow * 8 + ebase) = res;
  }
}

// ---------------------------------------------------------------------------
extern "C" void kernel_launch(void* const* d_in, const int* in_sizes, int n_in,
                              void* d_out, int out_size, void* d_ws, size_t ws_size,
                              hipStream_t stream) {
  (void)n_in; (void)out_size; (void)ws_size;
  const float* lstm = (const float*)d_in[0];
  const float* bot  = (const float*)d_in[1];
  const float* W1   = (const float*)d_in[2];
  const float* b1   = (const float*)d_in[3];
  const float* gam  = (const float*)d_in[4];
  const float* bet  = (const float*)d_in[5];
  const float* W2   = (const float*)d_in[6];
  const float* b2   = (const float*)d_in[7];
  const float* W3   = (const float*)d_in[8];
  const float* b3   = (const float*)d_in[9];
  float*    out = (float*)d_out;
  unsigned* ws  = (unsigned*)d_ws;   // needs 147456 B for packed bf16 fragments

  const int B = in_sizes[0] / 448;   // 262144

  gate_pack_weights<<<18, 256, 0, stream>>>(W1, W2, ws);
  const int nb = (B + 127) / 128;    // 128 rows per block (8 waves x 16 rows)
  gate_main<<<nb, 256, 0, stream>>>(lstm, bot, ws, b1, gam, bet, b2, W3, b3, out, B);
}
